// Pinard_34351148434151
// MI455X (gfx1250) — compile-verified, matrix operand order checked by reading, not run on hardware
//
#include <hip/hip_runtime.h>
#include <hip/hip_bf16.h>
#include <math.h>
#include <stdint.h>

#define D_IN    1024
#define N_BANKC 8192
#define M_ROWS  8192
#define D_OUT   1024

// GEMM block tiling: 32 rows x 512 cols, 256 threads (8 waves).
// K-step per staged tile: 64 elements (128 bytes of bf16 K-data per row).
// Each wave: 32x64 output = 2 A-frags x 4 B-frags x 2 K-chunks = 16 WMMA / tile.
#define LDS_STR 144                 // 128B K-data + 16B pad (bank spread)
#define ASZ     (32 * LDS_STR)      // 4608 B
#define BSZ     (512 * LDS_STR)     // 73728 B
#define BUFSZ   (ASZ + BSZ)         // 78336 B; x2 buffers = 153 KB LDS

typedef __attribute__((ext_vector_type(16))) __bf16 v16bf;
typedef __attribute__((ext_vector_type(8)))  __bf16 v8bf;
typedef __attribute__((ext_vector_type(4)))  __bf16 v4bf;
typedef __attribute__((ext_vector_type(8)))  float  v8f;

union BF16x16 { v16bf v; v8bf h[2]; };

// Double-buffered async->LDS WMMA mainloop (GVS-form async loads, ASYNCcnt).
// acc[u][t] = 16x16 f32 tile at rows u*16.., cols wave*64 + t*16..
__device__ __forceinline__ void gemm_pipeline(const __bf16* __restrict__ A,
                                              const __bf16* __restrict__ B, int K,
                                              int rowblock, int colblock,
                                              char* smem, v8f acc[2][4]) {
  const int tid  = threadIdx.x;
  const int lane = tid & 31;
  const int wave = tid >> 5;
  const int m = lane & 15, g = lane >> 4;

  // Loop-invariant per-thread address pieces.
  // B tile: 512 rows x 128B, b128 chunks: j=0..15, row = j*32 + tid/8, col16 = tid%8.
  // A tile:  32 rows x 128B, b64  chunks: s=0..1,  row = s*16 + tid/16, col8 = tid%16.
  const uint32_t voffB = (uint32_t)((tid >> 3) * K * 2 + (tid & 7) * 16);
  const uint32_t voffA = (uint32_t)((tid >> 4) * K * 2 + (tid & 15) * 8);
  const uint32_t smem0 = (uint32_t)(unsigned long long)smem;  // LDS byte offset
  uint32_t ldsB0[16], ldsA0[2];
#pragma unroll
  for (int j = 0; j < 16; ++j)
    ldsB0[j] = smem0 + ASZ + (uint32_t)((j * 32 + (tid >> 3)) * LDS_STR + (tid & 7) * 16);
#pragma unroll
  for (int s = 0; s < 2; ++s)
    ldsA0[s] = smem0 + (uint32_t)((s * 16 + (tid >> 4)) * LDS_STR + (tid & 15) * 8);

  const char* sB = (const char*)(B + (size_t)colblock * K);
  const char* sA = (const char*)(A + (size_t)rowblock * K);

  // Issue one K-step tile into buffer at LDS byte offset bufoff.
  // Exactly 18 async instructions per wave per buffer (uniform across waves).
  auto issue = [&](uint32_t bufoff, int k0bytes) {
#pragma unroll
    for (int j = 0; j < 16; ++j)
      asm volatile("global_load_async_to_lds_b128 %0, %1, %2"
                   :: "v"(ldsB0[j] + bufoff), "v"(voffB),
                      "s"(sB + (size_t)k0bytes + (size_t)j * 32 * K * 2)
                   : "memory");
#pragma unroll
    for (int s = 0; s < 2; ++s)
      asm volatile("global_load_async_to_lds_b64 %0, %1, %2"
                   :: "v"(ldsA0[s] + bufoff), "v"(voffA),
                      "s"(sA + (size_t)k0bytes + (size_t)s * 16 * K * 2)
                   : "memory");
  };

  auto compute = [&](uint32_t bufoff) {
    const char* cA = smem + bufoff;
    const char* cB = cA + ASZ;
#pragma unroll
    for (int ks = 0; ks < 2; ++ks) {          // two 32-wide K-chunks per tile
      BF16x16 af[2];
#pragma unroll
      for (int u = 0; u < 2; ++u) {
        const char* p = cA + (u * 16 + m) * LDS_STR + ks * 64 + g * 16;
        af[u].h[0] = *(const v8bf*)(p);
        af[u].h[1] = *(const v8bf*)(p + 32);
      }
#pragma unroll
      for (int t = 0; t < 4; ++t) {
        const char* p = cB + (wave * 64 + t * 16 + m) * LDS_STR + ks * 64 + g * 16;
        BF16x16 bfr;
        bfr.h[0] = *(const v8bf*)(p);
        bfr.h[1] = *(const v8bf*)(p + 32);
#pragma unroll
        for (int u = 0; u < 2; ++u)
          acc[u][t] = __builtin_amdgcn_wmma_f32_16x16x32_bf16(
              false, af[u].v, false, bfr.v, (short)0, acc[u][t], false, false);
      }
    }
  };

  const int NIT = K / 64;
  uint32_t curoff = 0;

  issue(0, 0);                                 // prologue: tile 0 -> buffer 0
  for (int it = 0; it < NIT - 1; ++it) {       // steady state (no branches inside)
    issue(curoff ^ BUFSZ, (it + 1) * 128);     // prefetch next tile
    // 18 cur-buffer ops done once <=18 remain outstanding (next 18 in flight).
    asm volatile("s_wait_asynccnt 0x12" ::: "memory");
    __syncthreads();
    compute(curoff);
    __syncthreads();                           // all waves done reading before refill
    curoff ^= BUFSZ;
  }
  asm volatile("s_wait_asynccnt 0x0" ::: "memory");  // peeled last tile
  __syncthreads();
  compute(curoff);
}

// ---------------- conversion kernels ----------------

__global__ __launch_bounds__(256) void cvt_f32_to_bf16(const float* __restrict__ in,
                                                       __bf16* __restrict__ out, int n4) {
  int gid = blockIdx.x * 256 + threadIdx.x;
  if (gid >= n4) return;
  const float4 f = ((const float4*)in)[gid];
  v4bf o;
  o[0] = (__bf16)f.x; o[1] = (__bf16)f.y; o[2] = (__bf16)f.z; o[3] = (__bf16)f.w;
  ((v4bf*)out)[gid] = o;
}

// v [N_BANK x D_OUT] f32 row-major -> vT [D_OUT x N_BANK] bf16 row-major
__global__ __launch_bounds__(256) void transpose_v_bf16(const float* __restrict__ v,
                                                        __bf16* __restrict__ vT) {
  __shared__ float tile[32][33];
  const int tx = threadIdx.x, ty = threadIdx.y;
  const int n0 = blockIdx.x * 32, o0 = blockIdx.y * 32;
#pragma unroll
  for (int j = 0; j < 32; j += 8)
    tile[ty + j][tx] = v[(size_t)(n0 + ty + j) * D_OUT + o0 + tx];
  __syncthreads();
#pragma unroll
  for (int j = 0; j < 32; j += 8)
    vT[(size_t)(o0 + ty + j) * N_BANKC + n0 + tx] = (__bf16)tile[tx][ty + j];
}

// ---------------- GEMM1: scores = q . k^T ----------------
__global__ __launch_bounds__(256) void gemm1_scores(const __bf16* __restrict__ qh,
                                                    const __bf16* __restrict__ kh,
                                                    __bf16* __restrict__ S,
                                                    float* __restrict__ partials) {
  __shared__ __align__(16) char smem[2 * BUFSZ];
  const int rowblock = blockIdx.y * 32;
  const int colblock = blockIdx.x * 512;

  v8f acc[2][4] = {};
  gemm_pipeline(qh, kh, D_IN, rowblock, colblock, smem, acc);

  const int lane = threadIdx.x & 31;
  const int wave = threadIdx.x >> 5;
  const int m = lane & 15, g = lane >> 4;
  const int colb = colblock + wave * 64;

#pragma unroll
  for (int u = 0; u < 2; ++u) {
#pragma unroll
    for (int r = 0; r < 8; ++r) {
      float s = acc[u][0][r] * acc[u][0][r] + acc[u][1][r] * acc[u][1][r] +
                acc[u][2][r] * acc[u][2][r] + acc[u][3][r] * acc[u][3][r];
      s += __shfl_xor(s, 1, 32);
      s += __shfl_xor(s, 2, 32);
      s += __shfl_xor(s, 4, 32);
      s += __shfl_xor(s, 8, 32);
      if (m == 0) {
        const int row = rowblock + u * 16 + r + 8 * g;
        partials[(size_t)row * (N_BANKC / 64) + (blockIdx.x * 8 + wave)] = s;
      }
    }
#pragma unroll
    for (int t = 0; t < 4; ++t)
#pragma unroll
      for (int r = 0; r < 8; ++r) {
        const int row = rowblock + u * 16 + r + 8 * g;
        S[(size_t)row * N_BANKC + colb + t * 16 + m] = (__bf16)acc[u][t][r];
      }
  }
}

// ---------------- per-row scale = sqrt(N) / ||row|| ----------------
__global__ __launch_bounds__(256) void row_scale(const float* __restrict__ partials,
                                                 float* __restrict__ scale) {
  const int lane = threadIdx.x & 31;
  const int wave = threadIdx.x >> 5;
  const int row  = blockIdx.x * 8 + wave;
  const float* p = partials + (size_t)row * (N_BANKC / 64);
  float s = p[lane] + p[lane + 32] + p[lane + 64] + p[lane + 96];
  s += __shfl_xor(s, 1, 32);
  s += __shfl_xor(s, 2, 32);
  s += __shfl_xor(s, 4, 32);
  s += __shfl_xor(s, 8, 32);
  s += __shfl_xor(s, 16, 32);
  if (lane == 0)
    scale[row] = sqrtf((float)N_BANKC) * rsqrtf(fmaxf(s, 1e-30f));
}

// ---------------- in-place exact GELU on bf16 scores ----------------
__global__ __launch_bounds__(256) void gelu_inplace(__bf16* __restrict__ S,
                                                    const float* __restrict__ scale) {
  const size_t gid  = (size_t)blockIdx.x * 256 + threadIdx.x;
  const size_t row  = gid >> 10;
  const size_t colc = (gid & 1023) << 3;
  const float sc = scale[row];
  v8bf* p = (v8bf*)(S + row * N_BANKC + colc);
  v8bf x = *p;
  v8bf y;
#pragma unroll
  for (int i = 0; i < 8; ++i) {
    const float v  = (float)x[i] * sc;
    const float gv = 0.5f * v * (1.0f + erff(v * 0.70710678118654752f));
    y[i] = (__bf16)gv;
  }
  *p = y;
}

// ---------------- GEMM2: out = gelu_scores . v (via vT) ----------------
__global__ __launch_bounds__(256) void gemm2_out(const __bf16* __restrict__ G,
                                                 const __bf16* __restrict__ vT,
                                                 float* __restrict__ out) {
  __shared__ __align__(16) char smem[2 * BUFSZ];
  const int rowblock = blockIdx.y * 32;
  const int colblock = blockIdx.x * 512;

  v8f acc[2][4] = {};
  gemm_pipeline(G, vT, N_BANKC, rowblock, colblock, smem, acc);

  const int lane = threadIdx.x & 31;
  const int wave = threadIdx.x >> 5;
  const int m = lane & 15, g = lane >> 4;
  const int colb = colblock + wave * 64;

#pragma unroll
  for (int u = 0; u < 2; ++u)
#pragma unroll
    for (int t = 0; t < 4; ++t)
#pragma unroll
      for (int r = 0; r < 8; ++r) {
        const int row = rowblock + u * 16 + r + 8 * g;
        out[(size_t)row * D_OUT + colb + t * 16 + m] = acc[u][t][r];
      }
}

// ---------------- host launcher ----------------
extern "C" void kernel_launch(void* const* d_in, const int* in_sizes, int n_in,
                              void* d_out, int out_size, void* d_ws, size_t ws_size,
                              hipStream_t stream) {
  (void)in_sizes; (void)n_in; (void)out_size; (void)ws_size;
  const float* q = (const float*)d_in[0];   // [4,2048,1024]
  const float* k = (const float*)d_in[1];   // [8192,1024]
  const float* v = (const float*)d_in[2];   // [8192,1024]
  float* out = (float*)d_out;               // [4,2048,1024]

  char* ws = (char*)d_ws;
  size_t off = 0;
  __bf16* qh = (__bf16*)(ws + off); off += (size_t)M_ROWS  * D_IN    * 2;
  __bf16* kh = (__bf16*)(ws + off); off += (size_t)N_BANKC * D_IN    * 2;
  __bf16* vT = (__bf16*)(ws + off); off += (size_t)D_OUT   * N_BANKC * 2;
  __bf16* S  = (__bf16*)(ws + off); off += (size_t)M_ROWS  * N_BANKC * 2;
  float* partials = (float*)(ws + off); off += (size_t)M_ROWS * (N_BANKC / 64) * 4;
  float* scale    = (float*)(ws + off); off += (size_t)M_ROWS * 4;

  const int n4 = M_ROWS * D_IN / 4;
  cvt_f32_to_bf16<<<(n4 + 255) / 256, 256, 0, stream>>>(q, qh, n4);
  cvt_f32_to_bf16<<<(n4 + 255) / 256, 256, 0, stream>>>(k, kh, n4);
  transpose_v_bf16<<<dim3(N_BANKC / 32, D_OUT / 32), dim3(32, 8), 0, stream>>>(v, vT);

  gemm1_scores<<<dim3(N_BANKC / 512, M_ROWS / 32), 256, 0, stream>>>(qh, kh, S, partials);
  row_scale<<<M_ROWS / 8, 256, 0, stream>>>(partials, scale);
  gelu_inplace<<<(M_ROWS * (N_BANKC / 8)) / 256, 256, 0, stream>>>(S, scale);
  gemm2_out<<<dim3(D_OUT / 512, M_ROWS / 32), 256, 0, stream>>>(S, vT, out);
}
